// WordInputCombiner_3092376453385
// MI455X (gfx1250) — compile-verified, matrix operand order checked by reading, not run on hardware
//
#include <hip/hip_runtime.h>

// ---------------------------------------------------------------------------
// WordInputCombiner forward for gfx1250 (MI455X).
// All GEMMs run on v_wmma_f32_16x16x32_bf16 (bf16 in, f32 accumulate) with
// double-buffered LDS staging + global_prefetch of the K-tile after next.
// ---------------------------------------------------------------------------

typedef __attribute__((ext_vector_type(16))) __bf16        v16bf;
typedef __attribute__((ext_vector_type(16))) unsigned short v16us;
typedef __attribute__((ext_vector_type(8)))  float          v8f;

#define D_MODEL 512
#define DFF     2048
#define NHEAD   8
#define DHEAD   64
#define NLAYER  2

#define BM 128
#define BN 128
#define BK 32
#define LDSK 40   // BK + 8 pad (bank-conflict mitigation); row = 80B, 8B aligned

__device__ __forceinline__ unsigned short f2bf(float f) {
  unsigned int u = __float_as_uint(f);
  u += 0x7FFFu + ((u >> 16) & 1u);   // round-to-nearest-even (finite inputs)
  return (unsigned short)(u >> 16);
}

// pack two floats into one dword of bf16 (lo -> [15:0], hi -> [31:16])
__device__ __forceinline__ unsigned int f2bf2(float a, float b) {
  unsigned int ua = __float_as_uint(a);
  unsigned int ub = __float_as_uint(b);
  ua += 0x7FFFu + ((ua >> 16) & 1u);
  ub += 0x7FFFu + ((ub >> 16) & 1u);
  return (ua >> 16) | (ub & 0xFFFF0000u);
}

__device__ __forceinline__ float blockReduceSum(float v, float* red) {
  #pragma unroll
  for (int o = 16; o > 0; o >>= 1) v += __shfl_xor(v, o, 32);
  const int lane = threadIdx.x & 31, wave = threadIdx.x >> 5;
  if (lane == 0) red[wave] = v;
  __syncthreads();
  if (wave == 0) {
    float t = (lane < ((int)blockDim.x >> 5)) ? red[lane] : 0.f;
    #pragma unroll
    for (int o = 4; o > 0; o >>= 1) t += __shfl_xor(t, o, 32);
    if (lane == 0) red[0] = t;
  }
  __syncthreads();
  float r = red[0];
  __syncthreads();
  return r;
}

__device__ __forceinline__ float blockReduceMax(float v, float* red) {
  #pragma unroll
  for (int o = 16; o > 0; o >>= 1) v = fmaxf(v, __shfl_xor(v, o, 32));
  const int lane = threadIdx.x & 31, wave = threadIdx.x >> 5;
  if (lane == 0) red[wave] = v;
  __syncthreads();
  if (wave == 0) {
    float t = (lane < ((int)blockDim.x >> 5)) ? red[lane] : -1e30f;
    #pragma unroll
    for (int o = 4; o > 0; o >>= 1) t = fmaxf(t, __shfl_xor(t, o, 32));
    if (lane == 0) red[0] = t;
  }
  __syncthreads();
  float r = red[0];
  __syncthreads();
  return r;
}

// ---------------------------------------------------------------------------
// Generic batched GEMM: C[z] = A[z] (MxK) * B[z] (KxN) [+ bias] [ReLU]
// BNK=true  -> B stored as N x K rows (used for Q*K^T scores)
// fp32 in memory, bf16 in LDS, f32 WMMA accumulation.
// Double-buffered staging: iteration i loads fragments from buffer cur while
// issuing tile i+1's global loads into buffer cur^1; one barrier/iteration.
// ---------------------------------------------------------------------------
template <bool BNK, bool RELU>
__global__ __launch_bounds__(256)
void gemm_kernel(const float* __restrict__ A, const float* __restrict__ B,
                 const float* __restrict__ bias, float* __restrict__ C,
                 int M, int N, int K, int lda, int ldb, int ldc, int H,
                 long long sAb, long long sAh, long long sBb, long long sBh,
                 long long sCb, long long sCh)
{
  __shared__ __align__(16) unsigned short As[2][BM][LDSK];
  __shared__ __align__(16) unsigned short Bs[2][BN][LDSK];

  const int z  = blockIdx.z;
  const int zb = z / H, zh = z % H;
  A += zb * sAb + zh * sAh;
  B += zb * sBb + zh * sBh;
  C += zb * sCb + zh * sCh;

  const int m0 = blockIdx.y * BM;
  const int n0 = blockIdx.x * BN;

  const int tid  = threadIdx.x;
  const int lane = tid & 31;
  const int wave = tid >> 5;
  const int wm   = (wave & 3) << 5;   // wave M offset: 0,32,64,96
  const int wn   = (wave >> 2) << 6;  // wave N offset: 0,64
  const int half = lane >> 4;         // 0/1 : lanes 0-15 vs 16-31
  const int l16  = lane & 15;

  const bool aFull = (m0 + BM <= M);
  const bool bFull = (n0 + BN <= N);

  const int kgrp = (tid & 7) << 2;  // 0..28 (4 consecutive k per thread)
  const int rrow = tid >> 3;        // 0..31
  const int kq   = tid >> 5;        // 0..7   (BKN staging)
  const int n4   = lane << 2;       // 0..124 (BKN staging)

  auto stageA = [&](int kt, int buf) {
    const bool kFull = (kt + BK <= K);
    if (aFull && kFull) {
      if (kt + 2 * BK <= K)   // prefetch the K-tile after the one being staged
        __builtin_prefetch(A + (long long)(m0 + rrow) * lda + (kt + BK + kgrp), 0, 0);
      #pragma unroll
      for (int mi = 0; mi < 4; ++mi) {
        const int m = rrow + (mi << 5);
        const float4 f = *reinterpret_cast<const float4*>(
            A + (long long)(m0 + m) * lda + (kt + kgrp));
        uint2 p; p.x = f2bf2(f.x, f.y); p.y = f2bf2(f.z, f.w);
        *reinterpret_cast<uint2*>(&As[buf][m][kgrp]) = p;
      }
    } else {
      #pragma unroll
      for (int mi = 0; mi < 4; ++mi) {
        const int m = rrow + (mi << 5);
        const long long gm  = (long long)m0 + m;
        const long long gmc = (gm < M) ? gm : (long long)(M - 1);
        float v[4];
        #pragma unroll
        for (int kk = 0; kk < 4; ++kk) {
          const int gk  = kt + kgrp + kk;
          const int gkc = (gk < K) ? gk : (K - 1);
          const float t = A[gmc * lda + gkc];       // always in-bounds
          v[kk] = (gm < M && gk < K) ? t : 0.f;     // branch-free select
        }
        uint2 p; p.x = f2bf2(v[0], v[1]); p.y = f2bf2(v[2], v[3]);
        *reinterpret_cast<uint2*>(&As[buf][m][kgrp]) = p;
      }
    }
  };

  auto stageB = [&](int kt, int buf) {
    const bool kFull = (kt + BK <= K);
    if (BNK) {
      if (bFull && kFull) {
        if (kt + 2 * BK <= K)
          __builtin_prefetch(B + (long long)(n0 + rrow) * ldb + (kt + BK + kgrp), 0, 0);
        #pragma unroll
        for (int ni = 0; ni < 4; ++ni) {
          const int n = rrow + (ni << 5);
          const float4 f = *reinterpret_cast<const float4*>(
              B + (long long)(n0 + n) * ldb + (kt + kgrp));
          uint2 p; p.x = f2bf2(f.x, f.y); p.y = f2bf2(f.z, f.w);
          *reinterpret_cast<uint2*>(&Bs[buf][n][kgrp]) = p;
        }
      } else {
        #pragma unroll
        for (int ni = 0; ni < 4; ++ni) {
          const int n = rrow + (ni << 5);
          const long long gn  = (long long)n0 + n;
          const long long gnc = (gn < N) ? gn : (long long)(N - 1);
          float v[4];
          #pragma unroll
          for (int kk = 0; kk < 4; ++kk) {
            const int gk  = kt + kgrp + kk;
            const int gkc = (gk < K) ? gk : (K - 1);
            const float t = B[gnc * ldb + gkc];
            v[kk] = (gn < N && gk < K) ? t : 0.f;
          }
          uint2 p; p.x = f2bf2(v[0], v[1]); p.y = f2bf2(v[2], v[3]);
          *reinterpret_cast<uint2*>(&Bs[buf][n][kgrp]) = p;
        }
      }
    } else {
      if (bFull && kFull) {
        if (kt + 2 * BK <= K)
          __builtin_prefetch(B + (long long)(kt + BK + kq) * ldb + (n0 + n4), 0, 0);
        #pragma unroll
        for (int ki = 0; ki < 4; ++ki) {
          const int k = kq + (ki << 3);
          const float4 f = *reinterpret_cast<const float4*>(
              B + (long long)(kt + k) * ldb + (n0 + n4));
          Bs[buf][n4 + 0][k] = f2bf(f.x);
          Bs[buf][n4 + 1][k] = f2bf(f.y);
          Bs[buf][n4 + 2][k] = f2bf(f.z);
          Bs[buf][n4 + 3][k] = f2bf(f.w);
        }
      } else {
        #pragma unroll
        for (int ki = 0; ki < 4; ++ki) {
          const int k   = kq + (ki << 3);
          const int gk  = kt + k;
          const int gkc = (gk < K) ? gk : (K - 1);
          #pragma unroll
          for (int nn = 0; nn < 4; ++nn) {
            const long long gn  = (long long)n0 + n4 + nn;
            const long long gnc = (gn < N) ? gn : (long long)(N - 1);
            const float t = B[(long long)gkc * ldb + gnc];
            Bs[buf][n4 + nn][k] = f2bf((gn < N && gk < K) ? t : 0.f);
          }
        }
      }
    }
  };

  v8f acc[2][4];
  #pragma unroll
  for (int mt = 0; mt < 2; ++mt)
    #pragma unroll
    for (int nt = 0; nt < 4; ++nt)
      #pragma unroll
      for (int r = 0; r < 8; ++r) acc[mt][nt][r] = 0.f;

  stageA(0, 0);
  stageB(0, 0);
  __syncthreads();

  int cur = 0;
  for (int kt = 0; kt < K; kt += BK) {
    // ---- fragment loads per the CDNA5 16-bit A/B VGPR layout ----
    // element e<8  -> K = 8*half + e        (VGPRs 0..3)
    // element e>=8 -> K = 16 + 8*half + e-8 (VGPRs 4..7)
    v16us au[2], bu[4];
    #pragma unroll
    for (int mt = 0; mt < 2; ++mt) {
      const int mr = wm + (mt << 4) + l16;
      #pragma unroll
      for (int e = 0; e < 8; ++e) {
        au[mt][e]     = As[cur][mr][(half << 3) + e];
        au[mt][e + 8] = As[cur][mr][16 + (half << 3) + e];
      }
    }
    #pragma unroll
    for (int nt = 0; nt < 4; ++nt) {
      const int nr = wn + (nt << 4) + l16;
      #pragma unroll
      for (int e = 0; e < 8; ++e) {
        bu[nt][e]     = Bs[cur][nr][(half << 3) + e];
        bu[nt][e + 8] = Bs[cur][nr][16 + (half << 3) + e];
      }
    }

    // ---- stage next tile into the other buffer while WMMAs run ----
    if (kt + BK < K) {
      stageA(kt + BK, cur ^ 1);
      stageB(kt + BK, cur ^ 1);
    }

    #pragma unroll
    for (int mt = 0; mt < 2; ++mt)
      #pragma unroll
      for (int nt = 0; nt < 4; ++nt)
        acc[mt][nt] = __builtin_amdgcn_wmma_f32_16x16x32_bf16(
            false, __builtin_bit_cast(v16bf, au[mt]),
            false, __builtin_bit_cast(v16bf, bu[nt]),
            (short)0, acc[mt][nt], false, false);

    __syncthreads();
    cur ^= 1;
  }

  // ---- epilogue: D layout lane->N (l16), element r -> M = 8*half + r ----
  if (aFull && bFull) {
    #pragma unroll
    for (int nt = 0; nt < 4; ++nt) {
      const int col = n0 + wn + (nt << 4) + l16;
      const float bv = bias ? bias[col] : 0.f;
      #pragma unroll
      for (int mt = 0; mt < 2; ++mt) {
        #pragma unroll
        for (int r = 0; r < 8; ++r) {
          const int row = m0 + wm + (mt << 4) + (half << 3) + r;
          float v = acc[mt][nt][r] + bv;
          if (RELU) v = fmaxf(v, 0.f);
          C[(long long)row * ldc + col] = v;
        }
      }
    }
  } else {
    #pragma unroll
    for (int nt = 0; nt < 4; ++nt) {
      const int col = n0 + wn + (nt << 4) + l16;
      if (col >= N) continue;
      const float bv = bias ? bias[col] : 0.f;
      #pragma unroll
      for (int mt = 0; mt < 2; ++mt) {
        #pragma unroll
        for (int r = 0; r < 8; ++r) {
          const int row = m0 + wm + (mt << 4) + (half << 3) + r;
          if (row >= M) continue;
          float v = acc[mt][nt][r] + bv;
          if (RELU) v = fmaxf(v, 0.f);
          C[(long long)row * ldc + col] = v;
        }
      }
    }
  }
}

// ---------------------------------------------------------------------------
// out = LN(x + sinusoid(pos)) * g + b, times (pos < len) mask.  D=512, 256 thr.
// ---------------------------------------------------------------------------
__global__ __launch_bounds__(256)
void posln_kernel(const float* __restrict__ x, const float* __restrict__ g,
                  const float* __restrict__ b, const int* __restrict__ lens,
                  float* __restrict__ out, int SL)
{
  __shared__ float red[32];
  const long long row = blockIdx.x;
  const int t = (int)(row % SL);
  const long long w = row / SL;
  const int len = lens[w];
  const float* xr = x + row * D_MODEL;
  float* orow = out + row * D_MODEL;
  const int tid = threadIdx.x;

  float v[2]; float s = 0.f;
  #pragma unroll
  for (int i = 0; i < 2; ++i) {
    const int d = tid + (i << 8);
    const int m = d >> 1;
    const float inv = __expf(-(float)(2 * m) * (9.210340372f / (float)D_MODEL));
    const float ang = (float)t * inv;
    const float pe  = (d & 1) ? __cosf(ang) : __sinf(ang);
    v[i] = xr[d] + pe;
    s += v[i];
  }
  const float mean = blockReduceSum(s, red) * (1.f / D_MODEL);
  float vs = 0.f;
  #pragma unroll
  for (int i = 0; i < 2; ++i) { const float d0 = v[i] - mean; vs += d0 * d0; }
  const float var  = blockReduceSum(vs, red) * (1.f / D_MODEL);
  const float rstd = rsqrtf(var + 1e-5f);
  const float mask = (t < len) ? 1.f : 0.f;
  #pragma unroll
  for (int i = 0; i < 2; ++i) {
    const int d = tid + (i << 8);
    orow[d] = ((v[i] - mean) * rstd * g[d] + b[d]) * mask;
  }
}

// out = LN(res + y) * g + b   (in-place safe: out may alias res)
__global__ __launch_bounds__(256)
void addln_kernel(const float* __restrict__ res, const float* __restrict__ y,
                  const float* __restrict__ g, const float* __restrict__ b,
                  float* __restrict__ out)
{
  __shared__ float red[32];
  const long long row = blockIdx.x;
  const float* rr = res + row * D_MODEL;
  const float* yr = y   + row * D_MODEL;
  float* orow = out + row * D_MODEL;
  const int tid = threadIdx.x;

  float v[2]; float s = 0.f;
  #pragma unroll
  for (int i = 0; i < 2; ++i) {
    const int d = tid + (i << 8);
    v[i] = rr[d] + yr[d];
    s += v[i];
  }
  const float mean = blockReduceSum(s, red) * (1.f / D_MODEL);
  float vs = 0.f;
  #pragma unroll
  for (int i = 0; i < 2; ++i) { const float d0 = v[i] - mean; vs += d0 * d0; }
  const float var  = blockReduceSum(vs, red) * (1.f / D_MODEL);
  const float rstd = rsqrtf(var + 1e-5f);
  #pragma unroll
  for (int i = 0; i < 2; ++i) {
    const int d = tid + (i << 8);
    orow[d] = (v[i] - mean) * rstd * g[d] + b[d];
  }
}

// In-place row softmax with scale + key-length mask (matches ref: -1e9 fill).
// Row index -> z = row/SL -> batch b = z/NHEAD -> len = lens[b].
__global__ __launch_bounds__(256)
void softmax_kernel(float* __restrict__ S, const int* __restrict__ lens,
                    int SL, int N, float scale)
{
  __shared__ float red[32];
  const long long row = blockIdx.x;
  const long long z = row / SL;
  const int b = (int)(z / NHEAD);
  const int len = lens[b];
  float* sr = S + row * (long long)N;
  const int tid = threadIdx.x;

  float mx = -1e30f;
  for (int j = tid; j < N; j += 256) {
    float v = sr[j] * scale;
    if (j >= len) v = -1e9f;
    sr[j] = v;
    mx = fmaxf(mx, v);
  }
  mx = blockReduceMax(mx, red);
  float sum = 0.f;
  for (int j = tid; j < N; j += 256) {
    const float e = __expf(sr[j] - mx);
    sr[j] = e;
    sum += e;
  }
  sum = blockReduceSum(sum, red);
  const float inv = 1.f / sum;
  for (int j = tid; j < N; j += 256) sr[j] *= inv;
}

// wi[w,p,:]: eos -> special[1]; p==0 -> special[0]; token -> enc[gidx]; else 0
__global__ __launch_bounds__(256)
void gather_kernel(const float* __restrict__ enc, const float* __restrict__ special,
                   const int* __restrict__ gidx, const unsigned char* __restrict__ tmask,
                   const unsigned char* __restrict__ emask,
                   float* __restrict__ wi, int maxwl)
{
  const long long row = blockIdx.x;
  const int p = (int)(row % maxwl);
  const int tid = threadIdx.x;
  const bool eos = emask[row] != 0;
  const bool tok = tmask[row] != 0;
  const long long src = gidx[row];
  #pragma unroll
  for (int i = 0; i < 2; ++i) {
    const int d = tid + (i << 8);
    float v;
    if (eos)          v = special[D_MODEL + d];
    else if (p == 0)  v = special[d];
    else if (tok)     v = enc[src * D_MODEL + d];
    else              v = 0.f;
    wi[row * D_MODEL + d] = v;
  }
}

__global__ __launch_bounds__(256)
void out0_kernel(const float* __restrict__ h, float* __restrict__ out, int maxwl)
{
  const long long w = blockIdx.x;
  const int tid = threadIdx.x;
  #pragma unroll
  for (int i = 0; i < 2; ++i) {
    const int d = tid + (i << 8);
    out[w * D_MODEL + d] = h[w * (long long)maxwl * D_MODEL + d];
  }
}

// ---------------------------------------------------------------------------
// Host side
// ---------------------------------------------------------------------------
template <bool BNK, bool RELU>
static void gemm(hipStream_t s,
                 const float* A, int lda, const float* B, int ldb,
                 const float* bias, float* C, int ldc,
                 int M, int N, int K,
                 int Z = 1, int H = 1,
                 long long sAb = 0, long long sAh = 0,
                 long long sBb = 0, long long sBh = 0,
                 long long sCb = 0, long long sCh = 0)
{
  dim3 grid((N + BN - 1) / BN, (M + BM - 1) / BM, Z);
  gemm_kernel<BNK, RELU><<<grid, 256, 0, s>>>(A, B, bias, C, M, N, K,
                                              lda, ldb, ldc, H,
                                              sAb, sAh, sBb, sBh, sCb, sCh);
}

struct EncBufs { float *h, *q, *k, *v, *o, *t, *big; };

// Param layout (per encoder, dict insertion order):
// 0 ln_emb_g 1 ln_emb_b 2 Wq 3 bq 4 Wk 5 bk 6 Wv 7 bv 8 Wo 9 bo
// 10 W1 11 b1 12 W2 13 b2 14 ln1_g 15 ln1_b 16 ln2_g 17 ln2_b
static void run_layers(const float* const* P, const EncBufs& B,
                       long long Rz, int BSz, int SLz, const int* lens,
                       hipStream_t s)
{
  const int D = D_MODEL;
  const float scale = 0.125f;  // dh^-0.5, dh=64
  for (int i = 0; i < NLAYER; ++i) {
    const float* Wq = P[2]  + (size_t)i * D * D;
    const float* bq = P[3]  + (size_t)i * D;
    const float* Wk = P[4]  + (size_t)i * D * D;
    const float* bk = P[5]  + (size_t)i * D;
    const float* Wv = P[6]  + (size_t)i * D * D;
    const float* bv = P[7]  + (size_t)i * D;
    const float* Wo = P[8]  + (size_t)i * D * D;
    const float* bo = P[9]  + (size_t)i * D;
    const float* W1 = P[10] + (size_t)i * D * DFF;
    const float* b1 = P[11] + (size_t)i * DFF;
    const float* W2 = P[12] + (size_t)i * DFF * D;
    const float* b2 = P[13] + (size_t)i * D;
    const float* l1g = P[14] + (size_t)i * D;
    const float* l1b = P[15] + (size_t)i * D;
    const float* l2g = P[16] + (size_t)i * D;
    const float* l2b = P[17] + (size_t)i * D;

    const int Mz = (int)Rz;
    // QKV projections
    gemm<false, false>(s, B.h, D, Wq, D, bq, B.q, D, Mz, D, D);
    gemm<false, false>(s, B.h, D, Wk, D, bk, B.k, D, Mz, D, D);
    gemm<false, false>(s, B.h, D, Wv, D, bv, B.v, D, Mz, D, D);
    // Scores S = Q * K^T, batched over (batch, head)
    gemm<true, false>(s, B.q, D, B.k, D, nullptr, B.big, SLz,
                      SLz, SLz, DHEAD, BSz * NHEAD, NHEAD,
                      (long long)SLz * D, DHEAD,
                      (long long)SLz * D, DHEAD,
                      (long long)NHEAD * SLz * SLz, (long long)SLz * SLz);
    softmax_kernel<<<(unsigned)((long long)BSz * NHEAD * SLz), 256, 0, s>>>(
        B.big, lens, SLz, SLz, scale);
    // O = P * V
    gemm<false, false>(s, B.big, SLz, B.v, D, nullptr, B.o, D,
                       SLz, DHEAD, SLz, BSz * NHEAD, NHEAD,
                       (long long)NHEAD * SLz * SLz, (long long)SLz * SLz,
                       (long long)SLz * D, DHEAD,
                       (long long)SLz * D, DHEAD);
    // Output projection + residual LN
    gemm<false, false>(s, B.o, D, Wo, D, bo, B.t, D, Mz, D, D);
    addln_kernel<<<(unsigned)Rz, 256, 0, s>>>(B.h, B.t, l1g, l1b, B.h);
    // FFN (mid reuses the scores buffer; lifetimes don't overlap)
    gemm<false, true >(s, B.h, D, W1, DFF, b1, B.big, DFF, Mz, DFF, D);
    gemm<false, false>(s, B.big, DFF, W2, D, b2, B.t, D, Mz, D, DFF);
    addln_kernel<<<(unsigned)Rz, 256, 0, s>>>(B.h, B.t, l2g, l2b, B.h);
  }
}

extern "C" void kernel_launch(void* const* d_in, const int* in_sizes, int n_in,
                              void* d_out, int out_size, void* d_ws, size_t ws_size,
                              hipStream_t stream)
{
  const float* x        = (const float*)d_in[0];
  const int*   lengths  = (const int*)d_in[1];
  const float* special  = (const float*)d_in[2];
  const float* const* ctx = (const float* const*)(d_in + 3);
  const float* const* wc  = (const float* const*)(d_in + 21);
  const int*           gidx  = (const int*)d_in[39];
  const unsigned char* tmask = (const unsigned char*)d_in[40];
  const unsigned char* emask = (const unsigned char*)d_in[41];
  const int*           wlens = (const int*)d_in[42];

  const int BS = 32, SL = 512;
  const int n_word = in_sizes[42];
  const int maxwl  = in_sizes[39] / n_word;

  const long long R1 = (long long)BS * SL;
  const long long R2 = (long long)n_word * maxwl;
  const long long R  = (R1 > R2) ? R1 : R2;

  // Workspace carve-up: 6 row buffers + one big buffer (scores / FFN-mid).
  char* ws = (char*)d_ws;
  size_t rowBytes = ((size_t)R * D_MODEL * sizeof(float) + 255) & ~(size_t)255;
  EncBufs B;
  B.h   = (float*)(ws + 0 * rowBytes);
  B.q   = (float*)(ws + 1 * rowBytes);
  B.k   = (float*)(ws + 2 * rowBytes);
  B.v   = (float*)(ws + 3 * rowBytes);
  B.o   = (float*)(ws + 4 * rowBytes);
  B.t   = (float*)(ws + 5 * rowBytes);
  B.big = (float*)(ws + 6 * rowBytes);

  // ---- Stage 1: context encoder over [32, 512, 512] ----
  posln_kernel<<<(unsigned)R1, 256, 0, stream>>>(x, ctx[0], ctx[1], lengths,
                                                 B.h, SL);
  run_layers(ctx, B, R1, BS, SL, lengths, stream);

  // ---- Stage 2: gather/scatter into word inputs, then word combiner ----
  gather_kernel<<<(unsigned)R2, 256, 0, stream>>>(B.h, special, gidx, tmask,
                                                  emask, B.t, maxwl);
  posln_kernel<<<(unsigned)R2, 256, 0, stream>>>(B.t, wc[0], wc[1], wlens,
                                                 B.h, maxwl);
  run_layers(wc, B, R2, n_word, maxwl, wlens, stream);

  // ---- Output: wenc[:, 0, :] ----
  out0_kernel<<<(unsigned)n_word, 256, 0, stream>>>(B.h, (float*)d_out, maxwl);
}
